// SWSA_26508538151244
// MI455X (gfx1250) — compile-verified, matrix operand order checked by reading, not run on hardware
//
#include <hip/hip_runtime.h>
#include <hip/hip_bf16.h>

// ---------------------------------------------------------------------------
// Problem constants (from reference): x [4,128,64,64]
// ---------------------------------------------------------------------------
#define BATCH 4
#define CDIM  128          // c
#define C2    256          // 2c
#define NTOK  4096         // h*w
#define NKC   (NTOK / 32)  // 128 K-chunks of 32 tokens
#define SCALE 0.08838834764831845f   // c^-0.5
#define SROW  4104         // padded LDS row stride (4096 + 8) in floats
#define QROW_OFF ((unsigned)(16 * SROW * sizeof(float)))   // LDS byte offset of qrow

typedef __attribute__((ext_vector_type(16))) _Float16 v16h;
typedef __attribute__((ext_vector_type(8)))  _Float16 v8h;
typedef __attribute__((ext_vector_type(8)))  float    v8f;

// K-half -> K index mapping for 16-bit WMMA A/B operands (CDNA5 ISA 7.12.2):
// lanes 0-15:  halves 0-7 -> K 0-7,  halves 8-15 -> K 16-23
// lanes 16-31: halves 0-7 -> K 8-15, halves 8-15 -> K 24-31
__device__ __forceinline__ int kmap(int hh, int lane) {
    return (hh & 7) + ((hh >> 3) << 4) + ((lane >> 4) << 3);
}

// ---------------------------------------------------------------------------
// Kernel 1: conv1x1 (c -> 2c) + BatchNorm (eval).
//   q: token-major f16  q[b][n][ch]                  (phase-1 operands)
//   v: WMMA-B tile-packed f16 vp[b][kc][g][lane][hh] (phase-3 B operands)
//      inverse of kmap: lane = cl + ((t>>3)&1)*16, hh = (t&7) + ((t>>4)&1)*8
// ---------------------------------------------------------------------------
__global__ void swsa_proj_qv(const float* __restrict__ x,
                             const float* __restrict__ w_inp,
                             const float* __restrict__ b_inp,
                             const float* __restrict__ bn_g,
                             const float* __restrict__ bn_b,
                             const float* __restrict__ bn_m,
                             const float* __restrict__ bn_v,
                             _Float16* __restrict__ q,
                             _Float16* __restrict__ vp) {
    int i  = blockIdx.x * blockDim.x + threadIdx.x;     // over BATCH*C2*NTOK
    int n  = i % NTOK;
    int o  = (i / NTOK) % C2;
    int bb = i / (NTOK * C2);

    const float* xr = x + (size_t)bb * CDIM * NTOK;     // x[bb][k][n]
    const float* wr = w_inp + (size_t)o * CDIM;

    float acc = 0.f;
#pragma unroll 8
    for (int k = 0; k < CDIM; ++k)
        acc = fmaf(wr[k], xr[(size_t)k * NTOK + n], acc);
    acc += b_inp[o];

    float sc = bn_g[o] * rsqrtf(bn_v[o] + 1e-5f);
    acc = acc * sc + (bn_b[o] - bn_m[o] * sc);

    _Float16 h = (_Float16)acc;
    if (o < CDIM) {
        q[((size_t)bb * NTOK + n) * CDIM + o] = h;
    } else {
        int ch = o - CDIM;
        int kc = n >> 5, t = n & 31;
        int g  = ch >> 4, cl = ch & 15;
        int ln = cl + (((t >> 3) & 1) << 4);
        int hh = (t & 7) + (((t >> 4) & 1) << 3);
        vp[((((size_t)bb * NKC + kc) * 8 + g) * 32 + ln) * 16 + hh] = h;
    }
}

// ---------------------------------------------------------------------------
// Kernel 2: per-workgroup 16-row attention stripe, fully resident in LDS.
//   phase 1: S = q qT * scale   (v_wmma_f32_16x16x32_f16, K=128 in 4 chunks)
//   phase 2: row softmax in LDS; stream atn to HBM; pack f16 probs in place
//   phase 3: y_stripe = atn_stripe @ v  (WMMA, K=4096, 2 accumulator chains)
//   phase 4: fused out-projection: out = W_out . y_stripe^T + b_out (WMMA)
// grid.x = BATCH * (NTOK/16), block = 256 threads (8 wave32)
// ---------------------------------------------------------------------------
__global__ void swsa_attn16(const _Float16* __restrict__ qf,
                            const _Float16* __restrict__ vpf,
                            const float* __restrict__ w_out,
                            const float* __restrict__ b_out,
                            float* __restrict__ atn_out,
                            float* __restrict__ out_y) {
    extern __shared__ char smem_raw[];
    float*    S    = (float*)smem_raw;                   // [16][SROW]
    _Float16* qrow = (_Float16*)(S + 16 * SROW);         // [16][CDIM] (reused ph4)

    const int tid  = threadIdx.x;
    const int wave = tid >> 5;
    const int lane = tid & 31;

    const int bb = blockIdx.x / (NTOK / 16);
    const int rb = (blockIdx.x % (NTOK / 16)) * 16;      // stripe row base

    const _Float16* qb = qf + (size_t)bb * NTOK * CDIM;

    // ---- async stage of the contiguous 4KB q stripe into LDS -------------
    // (GLOBAL_LOAD_ASYNC_TO_LDS_B128, tracked by ASYNCcnt; 256 thr * 16B)
    {
        const char* gsrc   = (const char*)(qb + (size_t)rb * CDIM) + tid * 16;
        unsigned    ldsoff = QROW_OFF + (unsigned)tid * 16;
        asm volatile("global_load_async_to_lds_b128 %0, %1, off"
                     :: "v"(ldsoff), "v"(gsrc) : "memory");
        asm volatile("s_wait_asynccnt 0x0" ::: "memory");
    }
    __syncthreads();

    // -------- phase 1: scores --------------------------------------------
    // A operand (the 16 stripe rows) is invariant across column tiles.
    v16h A[4];
#pragma unroll
    for (int kc = 0; kc < 4; ++kc)
#pragma unroll
        for (int hh = 0; hh < 16; ++hh)
            A[kc][hh] = qrow[(lane & 15) * CDIM + kc * 32 + kmap(hh, lane)];

    for (int ct = wave; ct < NTOK / 16; ct += 8) {       // 32 col tiles / wave
        const int colbase = ct * 16;
        v8f acc = {};
#pragma unroll
        for (int kc = 0; kc < 4; ++kc) {
            v16h Bm;
#pragma unroll
            for (int hh = 0; hh < 16; ++hh)
                Bm[hh] = qb[(size_t)(colbase + (lane & 15)) * CDIM
                            + kc * 32 + kmap(hh, lane)];
            acc = __builtin_amdgcn_wmma_f32_16x16x32_f16(
                      false, A[kc], false, Bm, (short)0, acc, false, false);
        }
        const int mhi = (lane >> 4) << 3;                // C/D: lanes>=16 hold M+8
        const int nn  = colbase + (lane & 15);
#pragma unroll
        for (int m = 0; m < 8; ++m)
            S[(m + mhi) * SROW + nn] = acc[m] * SCALE;
    }
    __syncthreads();

    // -------- phase 2: softmax over each 4096-wide row --------------------
    // Final pass also packs f16 probabilities in place into the low half of
    // each row's own f32 region (write to byte 2i always trails read of 4i).
    for (int r = wave; r < 16; r += 8) {                 // 2 rows per wave
        float*    row  = S + r * SROW;
        _Float16* hrow = (_Float16*)row;
        float mx = -3.0e38f;
        for (int i = lane; i < NTOK; i += 32) mx = fmaxf(mx, row[i]);
#pragma unroll
        for (int off = 16; off; off >>= 1) mx = fmaxf(mx, __shfl_xor(mx, off, 32));

        float sum = 0.f;
        for (int i = lane; i < NTOK; i += 32) {
            float e = __expf(row[i] - mx);
            row[i] = e;
            sum += e;
        }
#pragma unroll
        for (int off = 16; off; off >>= 1) sum += __shfl_xor(sum, off, 32);
        const float inv = 1.0f / sum;

        float* aout = atn_out + ((size_t)bb * NTOK + (rb + r)) * NTOK;
        for (int i = lane; i < NTOK; i += 32) {
            float p = row[i] * inv;
            aout[i] = p;                                  // coalesced HBM write
            hrow[i] = (_Float16)p;                        // in-place f16 pack
        }
    }
    __syncthreads();

    // -------- phase 3: y_stripe = atn_stripe @ v --------------------------
    // each wave owns one 16-channel tile; K = 4096 in 128 steps, 2 acc chains.
    v8f acc0 = {}, acc1 = {};
    {
        // A: f16 probs, two contiguous 16B runs per lane from the LDS stripe
        const _Float16* arow = (const _Float16*)(S + (lane & 15) * SROW)
                             + ((lane >> 4) << 3);
        // B: tile-packed v -> one contiguous 32B v16h per lane per step
        const v16h* vpb = (const v16h*)vpf
                        + (((size_t)bb * NKC) * 8 + wave) * 32 + lane;

        for (int kc = 0; kc < NKC; kc += 2) {
#pragma unroll
            for (int u = 0; u < 2; ++u) {
                const int kb = (kc + u) * 32;
                v8h lo = *(const v8h*)(arow + kb);        // K: base+0..7
                v8h hi = *(const v8h*)(arow + kb + 16);   // K: base+16..23
                v16h Am = __builtin_shufflevector(lo, hi,
                              0, 1, 2, 3, 4, 5, 6, 7,
                              8, 9, 10, 11, 12, 13, 14, 15);
                v16h Bm = vpb[(size_t)(kc + u) * 8 * 32];
                if (u == 0)
                    acc0 = __builtin_amdgcn_wmma_f32_16x16x32_f16(
                               false, Am, false, Bm, (short)0, acc0, false, false);
                else
                    acc1 = __builtin_amdgcn_wmma_f32_16x16x32_f16(
                               false, Am, false, Bm, (short)0, acc1, false, false);
            }
            if (kc + 4 < NKC)                            // stream-ahead for v
                __builtin_prefetch(&vpb[(size_t)(kc + 4) * 8 * 32], 0, 1);
        }
    }
    __syncthreads();     // qrow reuse below

    // stage y_stripe (16 tokens x 128 ch) as f16 into qrow for phase 4
    {
        const int mhi = (lane >> 4) << 3;
        const int cb  = wave * 16;
#pragma unroll
        for (int m = 0; m < 8; ++m)
            qrow[(m + mhi) * CDIM + cb + (lane & 15)]
                = (_Float16)(acc0[m] + acc1[m]);
    }
    __syncthreads();

    // -------- phase 4: out projection, one 16x16 tile per wave ------------
    // out[token][o] = sum_c y[token][c] * w_out[o][c] + b_out[o]
    {
        const int ob = wave * 16;                        // out-channel base
        const float* wrow = w_out + (size_t)(ob + (lane & 15)) * CDIM;
        v8f accY = {};
#pragma unroll
        for (int kc = 0; kc < 4; ++kc) {
            v16h Am, Bm;
#pragma unroll
            for (int hh = 0; hh < 16; ++hh) {
                const int k = kc * 32 + kmap(hh, lane);
                Am[hh] = qrow[(lane & 15) * CDIM + k];   // y columns as A rows
                Bm[hh] = (_Float16)wrow[k];              // w_out row as B col
            }
            accY = __builtin_amdgcn_wmma_f32_16x16x32_f16(
                       false, Am, false, Bm, (short)0, accY, false, false);
        }
        const int   mhi  = (lane >> 4) << 3;
        const float bias = b_out[ob + (lane & 15)];
        // out_y[b][o][token]: per lane 8 consecutive tokens -> 32B runs
        float* yo = out_y + ((size_t)bb * CDIM + ob + (lane & 15)) * NTOK
                  + rb + mhi;
#pragma unroll
        for (int m = 0; m < 8; ++m)
            yo[m] = accY[m] + bias;
    }
}

// ---------------------------------------------------------------------------
extern "C" void kernel_launch(void* const* d_in, const int* in_sizes, int n_in,
                              void* d_out, int out_size, void* d_ws, size_t ws_size,
                              hipStream_t stream) {
    const float* x      = (const float*)d_in[0];
    const float* w_inp  = (const float*)d_in[1];
    const float* b_inp  = (const float*)d_in[2];
    const float* bn_g   = (const float*)d_in[3];
    const float* bn_b   = (const float*)d_in[4];
    const float* bn_m   = (const float*)d_in[5];
    const float* bn_v   = (const float*)d_in[6];
    const float* w_out  = (const float*)d_in[7];
    const float* b_out  = (const float*)d_in[8];

    float* out_y   = (float*)d_out;                               // [4,128,64,64]
    float* out_atn = out_y + (size_t)BATCH * CDIM * NTOK;         // [4,4096,4096]

    // workspace: q_f16 (4MB) | v_f16 packed (4MB)
    _Float16* q  = (_Float16*)d_ws;
    _Float16* vp = q + (size_t)BATCH * NTOK * CDIM;

    // 1) fused projection + BN -> f16 q (token-major) / v (B-tile-packed)
    swsa_proj_qv<<<(BATCH * C2 * NTOK) / 256, 256, 0, stream>>>(
        x, w_inp, b_inp, bn_g, bn_b, bn_m, bn_v, q, vp);

    // 2) attention stripes + fused out projection:
    //    267KB dynamic LDS per WGP (CDNA5: 320KB available)
    const size_t smem = (size_t)16 * SROW * sizeof(float)
                      + (size_t)16 * CDIM * sizeof(_Float16);
    hipFuncSetAttribute(reinterpret_cast<const void*>(swsa_attn16),
                        hipFuncAttributeMaxDynamicSharedMemorySize, (int)smem);
    swsa_attn16<<<BATCH * (NTOK / 16), 256, smem, stream>>>(
        q, vp, w_out, b_out, out_atn, out_y);
}
